// AttentionCommModule_84035330113931
// MI455X (gfx1250) — matrix operand order repaired and checked by reading, not compiled, not run on hardware
//
#include <hip/hip_runtime.h>
#include <hip/hip_bf16.h>

// Problem constants (match reference)
#define BB   512
#define NN   128
#define OBSD 128
#define MSGD 64
#define NH   4
#define DHD  16
#define EH   128
#define IH   256

typedef __bf16 bf16_t;
typedef __attribute__((ext_vector_type(16))) __bf16 v16bf;
typedef __attribute__((ext_vector_type(8)))  float  v8f;

__device__ __forceinline__ bf16_t f2bf(float x) { return (bf16_t)x; }

// A-matrix fragment: 16(M) x 32(K) bf16, source row-major [M][K] with `stride`.
// ISA 7.12.2: lanes 0-15 -> M=lane, K in {0..7, 16..23}; lanes 16-31 -> +8 K offset.
__device__ __forceinline__ v16bf load_a_frag(const bf16_t* base, int stride, int lane) {
  const int m  = lane & 15;
  const int kb = (lane >> 4) << 3;
  const bf16_t* rp = base + m * stride;
  v16bf a;
#pragma unroll
  for (int j = 0; j < 8; ++j) {
    const int k = ((j >> 2) << 4) + kb + ((j & 3) << 1);
    a[2 * j]     = rp[k];
    a[2 * j + 1] = rp[k + 1];
  }
  return a;
}

// B-matrix fragment: 32(K) x 16(N) bf16, source is B^T row-major [N][K] with `stride`.
// lanes 0-15 -> N=lane, K in {0..7, 16..23}; lanes 16-31 -> +8 K offset.
__device__ __forceinline__ v16bf load_b_frag(const bf16_t* base, int stride, int lane) {
  const int n  = lane & 15;
  const int kb = (lane >> 4) << 3;
  const bf16_t* rp = base + n * stride;
  v16bf b;
#pragma unroll
  for (int j = 0; j < 8; ++j) {
    const int k = ((j >> 2) << 4) + kb + ((j & 3) << 1);
    b[2 * j]     = rp[k];
    b[2 * j + 1] = rp[k + 1];
  }
  return b;
}

__device__ __forceinline__ v8f bf_wmma(v16bf a, v16bf b, v8f c) {
  return __builtin_amdgcn_wmma_f32_16x16x32_bf16(false, a, false, b, (short)0, c,
                                                 false, false);
}

// ---------------------------------------------------------------------------
// Kernel 1: message encoder MLP.  One block = 128 tokens, 8 waves x 16 rows.
// msgs = relu(obs @ w1 + b1) @ w2 + b2 ; write f32 msgs output + bf16 to ws.
// ---------------------------------------------------------------------------
__global__ __launch_bounds__(256) void k_encoder(
    const float* __restrict__ obs, const float* __restrict__ w1,
    const float* __restrict__ b1, const float* __restrict__ w2,
    const float* __restrict__ b2, float* __restrict__ out_msgs,
    bf16_t* __restrict__ ws_msgs) {
  extern __shared__ char smem[];
  bf16_t* w1T = (bf16_t*)smem;            // [EH=128][OBSD=128]
  bf16_t* w2T = w1T + EH * OBSD;          // [MSGD=64][EH=128]
  bf16_t* aob = w2T + MSGD * EH;          // [128 tok][OBSD]
  bf16_t* h1  = aob + NN * OBSD;          // [128 tok][EH]

  const int tid    = threadIdx.x;
  const int token0 = blockIdx.x * NN;

  for (int idx = tid; idx < EH * OBSD; idx += 256) {
    const int o = idx >> 7, i = idx & 127;           // w1T[o][i] = w1[i][o]
    w1T[idx] = f2bf(w1[i * EH + o]);
  }
  for (int idx = tid; idx < MSGD * EH; idx += 256) {
    const int o = idx >> 7, i = idx & 127;           // w2T[o][i] = w2[i][o]
    w2T[idx] = f2bf(w2[i * MSGD + o]);
  }
  for (int idx = tid; idx < NN * OBSD; idx += 256)
    aob[idx] = f2bf(obs[(size_t)token0 * OBSD + idx]);
  __syncthreads();

  const int wave = tid >> 5, lane = tid & 31;
  const int mrow = wave * 16;
  const int cn   = lane & 15;
  const int cm0  = (lane >> 4) << 3;

  // GEMM1: (16x128) @ (128x128) + bias, ReLU -> h1 (bf16).  A-frags hoisted.
  v16bf va1[4];
#pragma unroll
  for (int kt = 0; kt < 4; ++kt)
    va1[kt] = load_a_frag(aob + mrow * OBSD + kt * 32, OBSD, lane);
#pragma unroll
  for (int nt = 0; nt < 8; ++nt) {
    v8f acc = {};
#pragma unroll
    for (int kt = 0; kt < 4; ++kt)
      acc = bf_wmma(va1[kt], load_b_frag(w1T + (nt * 16) * OBSD + kt * 32, OBSD, lane), acc);
    const float bias = b1[nt * 16 + cn];
#pragma unroll
    for (int r = 0; r < 8; ++r) {
      float v = acc[r] + bias;
      v = v > 0.f ? v : 0.f;
      h1[(mrow + cm0 + r) * EH + nt * 16 + cn] = f2bf(v);
    }
  }
  // h1 rows are wave-local (written and read only by this wave) -> no barrier.

  // GEMM2: (16x128) @ (128x64) + bias -> msgs.  A-frags hoisted.
  v16bf va2[4];
#pragma unroll
  for (int kt = 0; kt < 4; ++kt)
    va2[kt] = load_a_frag(h1 + mrow * EH + kt * 32, EH, lane);
#pragma unroll
  for (int nt = 0; nt < 4; ++nt) {
    v8f acc = {};
#pragma unroll
    for (int kt = 0; kt < 4; ++kt)
      acc = bf_wmma(va2[kt], load_b_frag(w2T + (nt * 16) * EH + kt * 32, EH, lane), acc);
    const float bias = b2[nt * 16 + cn];
#pragma unroll
    for (int r = 0; r < 8; ++r) {
      const float v = acc[r] + bias;
      const int t = token0 + mrow + cm0 + r;
      const int c = nt * 16 + cn;
      out_msgs[(size_t)t * MSGD + c] = v;
      ws_msgs[(size_t)t * MSGD + c] = f2bf(v);
    }
  }
}

// ---------------------------------------------------------------------------
// Kernel 2: 4-head self-attention over the 128 agent tokens of one batch.
// One block = one batch.  Head dim 16 zero-padded to 32 so each 16x16 score
// tile is a single bf16 WMMA.  Softmax stats via xor-shuffle inside the
// 16-lane C-matrix halves; 1/rowsum folded in after the ctx GEMM.
// ---------------------------------------------------------------------------
__global__ __launch_bounds__(256) void k_attn(
    const float* __restrict__ wq, const float* __restrict__ bq,
    const float* __restrict__ wk, const float* __restrict__ bk,
    const float* __restrict__ wv, const float* __restrict__ bv,
    const float* __restrict__ wo, const float* __restrict__ bo,
    const bf16_t* __restrict__ ws_msgs, bf16_t* __restrict__ ws_agg) {
  extern __shared__ char smem[];
  bf16_t* wqT   = (bf16_t*)smem;             // [64][64]
  bf16_t* wkT   = wqT + MSGD * MSGD;
  bf16_t* wvT   = wkT + MSGD * MSGD;
  bf16_t* woT   = wvT + MSGD * MSGD;
  bf16_t* mb    = woT + MSGD * MSGD;         // [128][64] msgs bf16
  bf16_t* qp    = mb + NN * MSGD;            // [4][128][32] (padded head dim)
  bf16_t* kp    = qp + NH * NN * 32;         // [4][128][32]
  bf16_t* vT    = kp + NH * NN * 32;         // [64 dim][128 tok]
  bf16_t* ctxb  = vT + MSGD * NN;            // [128][64]
  bf16_t* attnb = ctxb + NN * MSGD;          // [8 waves][16][128]

  const int tid    = threadIdx.x;
  const int token0 = blockIdx.x * NN;

  for (int idx = tid; idx < MSGD * MSGD; idx += 256) {
    const int o = idx >> 6, i = idx & 63;
    wqT[idx] = f2bf(wq[i * MSGD + o]);
    wkT[idx] = f2bf(wk[i * MSGD + o]);
    wvT[idx] = f2bf(wv[i * MSGD + o]);
    woT[idx] = f2bf(wo[i * MSGD + o]);
  }
  for (int idx = tid; idx < NN * MSGD; idx += 256)
    mb[idx] = ws_msgs[(size_t)token0 * MSGD + idx];
  // zero the K-padding halves of qp / kp
  for (int idx = tid; idx < NH * NN * 16; idx += 256) {
    const int ht = idx >> 4, j = idx & 15;   // ht = h*128 + token
    qp[ht * 32 + 16 + j] = f2bf(0.f);
    kp[ht * 32 + 16 + j] = f2bf(0.f);
  }
  __syncthreads();

  const int wave = tid >> 5, lane = tid & 31;
  const int mrow = wave * 16;
  const int cn   = lane & 15;
  const int cm0  = (lane >> 4) << 3;

  // QKV projections for this wave's 16 tokens: nt tile == head index.
  // Shared A-frags (the wave's msgs rows) hoisted across heads & projections.
  v16bf vam[2];
#pragma unroll
  for (int kt = 0; kt < 2; ++kt)
    vam[kt] = load_a_frag(mb + mrow * MSGD + kt * 32, MSGD, lane);
#pragma unroll
  for (int h = 0; h < NH; ++h) {
    v8f aq = {}, ak = {}, av = {};
#pragma unroll
    for (int kt = 0; kt < 2; ++kt) {
      aq = bf_wmma(vam[kt], load_b_frag(wqT + (h * 16) * MSGD + kt * 32, MSGD, lane), aq);
      ak = bf_wmma(vam[kt], load_b_frag(wkT + (h * 16) * MSGD + kt * 32, MSGD, lane), ak);
      av = bf_wmma(vam[kt], load_b_frag(wvT + (h * 16) * MSGD + kt * 32, MSGD, lane), av);
    }
    const float biq = bq[h * 16 + cn], bik = bk[h * 16 + cn], biv = bv[h * 16 + cn];
#pragma unroll
    for (int r = 0; r < 8; ++r) {
      const int t = mrow + cm0 + r;
      qp[(h * NN + t) * 32 + cn] = f2bf(aq[r] + biq);
      kp[(h * NN + t) * 32 + cn] = f2bf(ak[r] + bik);
      vT[(h * 16 + cn) * NN + t] = f2bf(av[r] + biv);
    }
  }
  __syncthreads();  // scores need every token's k / v

  bf16_t* arow = attnb + wave * 16 * NN;
#pragma unroll
  for (int h = 0; h < NH; ++h) {
    v16bf aq = load_a_frag(qp + h * NN * 32 + mrow * 32, 32, lane);
    v8f sf[8];
#pragma unroll
    for (int ktile = 0; ktile < 8; ++ktile) {
      v8f z = {};
      sf[ktile] = bf_wmma(aq, load_b_frag(kp + h * NN * 32 + (ktile * 16) * 32, 32, lane), z);
    }
    float mx[8], sm[8];
#pragma unroll
    for (int r = 0; r < 8; ++r) mx[r] = -3.0e38f;
#pragma unroll
    for (int kt = 0; kt < 8; ++kt)
#pragma unroll
      for (int r = 0; r < 8; ++r) {
        const float s = sf[kt][r] * 0.25f;  // 1/sqrt(DH)
        sf[kt][r] = s;
        mx[r] = fmaxf(mx[r], s);
      }
#pragma unroll
    for (int off = 8; off >= 1; off >>= 1)
#pragma unroll
      for (int r = 0; r < 8; ++r) mx[r] = fmaxf(mx[r], __shfl_xor(mx[r], off, 32));
#pragma unroll
    for (int r = 0; r < 8; ++r) sm[r] = 0.f;
#pragma unroll
    for (int kt = 0; kt < 8; ++kt)
#pragma unroll
      for (int r = 0; r < 8; ++r) {
        const float e = __expf(sf[kt][r] - mx[r]);
        sm[r] += e;
        arow[(cm0 + r) * NN + kt * 16 + cn] = f2bf(e);  // C-layout -> A-layout transpose
      }
#pragma unroll
    for (int off = 8; off >= 1; off >>= 1)
#pragma unroll
      for (int r = 0; r < 8; ++r) sm[r] += __shfl_xor(sm[r], off, 32);

    v8f cx = {};
#pragma unroll
    for (int kt = 0; kt < 4; ++kt) {
      v16bf aa = load_a_frag(arow + kt * 32, NN, lane);
      cx = bf_wmma(aa, load_b_frag(vT + (h * 16) * NN + kt * 32, NN, lane), cx);
    }
#pragma unroll
    for (int r = 0; r < 8; ++r)
      ctxb[(mrow + cm0 + r) * MSGD + h * 16 + cn] = f2bf(cx[r] / sm[r]);
  }

  // output projection -> agg (bf16 workspace).  A-frags hoisted.
  v16bf vac[2];
#pragma unroll
  for (int kt = 0; kt < 2; ++kt)
    vac[kt] = load_a_frag(ctxb + mrow * MSGD + kt * 32, MSGD, lane);
#pragma unroll
  for (int nt = 0; nt < 4; ++nt) {
    v8f acc = {};
#pragma unroll
    for (int kt = 0; kt < 2; ++kt)
      acc = bf_wmma(vac[kt], load_b_frag(woT + (nt * 16) * MSGD + kt * 32, MSGD, lane), acc);
    const float bias = bo[nt * 16 + cn];
#pragma unroll
    for (int r = 0; r < 8; ++r) {
      const int t = token0 + mrow + cm0 + r;
      ws_agg[(size_t)t * MSGD + nt * 16 + cn] = f2bf(acc[r] + bias);
    }
  }
}

// ---------------------------------------------------------------------------
// Kernel 3: integrator MLP with LayerNorm.  One block = 128 tokens.
// h = concat(obs,agg) @ W1 + b1 ; LN stats via xor-shuffle over the C-frags;
// ReLU ; out = h @ W2 + b2.
// ---------------------------------------------------------------------------
__global__ __launch_bounds__(256) void k_integrator(
    const float* __restrict__ obs, const float* __restrict__ w1,
    const float* __restrict__ b1, const float* __restrict__ lng,
    const float* __restrict__ lnb, const float* __restrict__ w2,
    const float* __restrict__ b2, const bf16_t* __restrict__ ws_agg,
    float* __restrict__ out) {
  extern __shared__ char smem[];
  const int KC = OBSD + MSGD;                // 192
  bf16_t* w1T = (bf16_t*)smem;               // [IH=256][192]
  bf16_t* w2T = w1T + IH * KC;               // [OBSD=128][IH=256]
  bf16_t* ac  = w2T + OBSD * IH;             // [128 tok][192]
  bf16_t* hb  = ac + NN * KC;                // [128 tok][256]

  const int tid    = threadIdx.x;
  const int token0 = blockIdx.x * NN;

  for (int o = tid; o < IH; o += 256)        // transpose w1 -> [out][in]
    for (int i = 0; i < KC; ++i) w1T[o * KC + i] = f2bf(w1[i * IH + o]);
  for (int o = tid; o < OBSD; o += 256)      // transpose w2 -> [out][in]
    for (int i = 0; i < IH; ++i) w2T[o * IH + i] = f2bf(w2[i * OBSD + o]);
  for (int idx = tid; idx < NN * OBSD; idx += 256) {
    const int r = idx >> 7, c = idx & 127;
    ac[r * KC + c] = f2bf(obs[((size_t)token0 + r) * OBSD + c]);
  }
  for (int idx = tid; idx < NN * MSGD; idx += 256) {
    const int r = idx >> 6, c = idx & 63;
    ac[r * KC + OBSD + c] = ws_agg[((size_t)token0 + r) * MSGD + c];
  }
  __syncthreads();

  const int wave = tid >> 5, lane = tid & 31;
  const int mrow = wave * 16;
  const int cn   = lane & 15;
  const int cm0  = (lane >> 4) << 3;

  // GEMM1: (16x192) @ (192x256) + b1 -> acc[16] C-fragments
  v16bf va[6];
#pragma unroll
  for (int kt = 0; kt < 6; ++kt)
    va[kt] = load_a_frag(ac + mrow * KC + kt * 32, KC, lane);
  v8f acc[16];
#pragma unroll
  for (int nt = 0; nt < 16; ++nt) {
    v8f c = {};
#pragma unroll
    for (int kt = 0; kt < 6; ++kt)
      c = bf_wmma(va[kt], load_b_frag(w1T + (nt * 16) * KC + kt * 32, KC, lane), c);
    const float bias = b1[nt * 16 + cn];
#pragma unroll
    for (int r = 0; r < 8; ++r) c[r] += bias;
    acc[nt] = c;
  }

  // LayerNorm over the 256 columns of each row, entirely in registers
  float s[8], sq[8];
#pragma unroll
  for (int r = 0; r < 8; ++r) { s[r] = 0.f; sq[r] = 0.f; }
#pragma unroll
  for (int nt = 0; nt < 16; ++nt)
#pragma unroll
    for (int r = 0; r < 8; ++r) {
      const float v = acc[nt][r];
      s[r] += v;
      sq[r] += v * v;
    }
#pragma unroll
  for (int off = 8; off >= 1; off >>= 1)
#pragma unroll
    for (int r = 0; r < 8; ++r) {
      s[r]  += __shfl_xor(s[r], off, 32);
      sq[r] += __shfl_xor(sq[r], off, 32);
    }
  float mean[8], rstd[8];
#pragma unroll
  for (int r = 0; r < 8; ++r) {
    mean[r] = s[r] * (1.f / IH);
    const float var = sq[r] * (1.f / IH) - mean[r] * mean[r];
    rstd[r] = rsqrtf(var + 1e-5f);
  }
#pragma unroll
  for (int nt = 0; nt < 16; ++nt) {
    const float gg = lng[nt * 16 + cn], bb = lnb[nt * 16 + cn];
#pragma unroll
    for (int r = 0; r < 8; ++r) {
      float v = (acc[nt][r] - mean[r]) * rstd[r] * gg + bb;
      v = v > 0.f ? v : 0.f;
      hb[(mrow + cm0 + r) * IH + nt * 16 + cn] = f2bf(v);
    }
  }

  // GEMM2: (16x256) @ (256x128) + b2 -> enriched (f32 output)
  v16bf va2[8];
#pragma unroll
  for (int kt = 0; kt < 8; ++kt)
    va2[kt] = load_a_frag(hb + mrow * IH + kt * 32, IH, lane);
#pragma unroll
  for (int nt = 0; nt < 8; ++nt) {
    v8f c = {};
#pragma unroll
    for (int kt = 0; kt < 8; ++kt)
      c = bf_wmma(va2[kt], load_b_frag(w2T + (nt * 16) * IH + kt * 32, IH, lane), c);
    const float bias = b2[nt * 16 + cn];
#pragma unroll
    for (int r = 0; r < 8; ++r)
      out[((size_t)token0 + mrow + cm0 + r) * OBSD + nt * 16 + cn] = c[r] + bias;
  }
}

// ---------------------------------------------------------------------------
extern "C" void kernel_launch(void* const* d_in, const int* in_sizes, int n_in,
                              void* d_out, int out_size, void* d_ws, size_t ws_size,
                              hipStream_t stream) {
  const float* obs    = (const float*)d_in[0];
  const float* enc_w1 = (const float*)d_in[1];
  const float* enc_b1 = (const float*)d_in[2];
  const float* enc_w2 = (const float*)d_in[3];
  const float* enc_b2 = (const float*)d_in[4];
  const float* wq = (const float*)d_in[5];  const float* bq = (const float*)d_in[6];
  const float* wk = (const float*)d_in[7];  const float* bk = (const float*)d_in[8];
  const float* wv = (const float*)d_in[9];  const float* bv = (const float*)d_in[10];
  const float* wo = (const float*)d_in[11]; const float* bo = (const float*)d_in[12];
  const float* int_w1 = (const float*)d_in[13];
  const float* int_b1 = (const float*)d_in[14];
  const float* ln_g   = (const float*)d_in[15];
  const float* ln_b   = (const float*)d_in[16];
  const float* int_w2 = (const float*)d_in[17];
  const float* int_b2 = (const float*)d_in[18];

  float* out_enriched = (float*)d_out;                               // B*N*128
  float* out_msgs     = out_enriched + (size_t)BB * NN * OBSD;       // B*N*64
  bf16_t* ws_msgs = (bf16_t*)d_ws;                                   // B*N*64 bf16
  bf16_t* ws_agg  = ws_msgs + (size_t)BB * NN * MSGD;                // B*N*64 bf16

  constexpr size_t LDS1 = (size_t)(128 * 128 + 64 * 128 + 128 * 128 + 128 * 128) * sizeof(bf16_t);
  constexpr size_t LDS2 = (size_t)(4 * 64 * 64 + 128 * 64 + 2 * 4 * 128 * 32 +
                                   64 * 128 + 128 * 64 + 8 * 16 * 128) * sizeof(bf16_t);
  constexpr size_t LDS3 = (size_t)(256 * 192 + 128 * 256 + 128 * 192 + 128 * 256) * sizeof(bf16_t);

  (void)hipFuncSetAttribute((const void*)k_encoder,
                            hipFuncAttributeMaxDynamicSharedMemorySize, (int)LDS1);
  (void)hipFuncSetAttribute((const void*)k_attn,
                            hipFuncAttributeMaxDynamicSharedMemorySize, (int)LDS2);
  (void)hipFuncSetAttribute((const void*)k_integrator,
                            hipFuncAttributeMaxDynamicSharedMemorySize, (int)LDS3);

  k_encoder<<<BB * NN / NN, 256, LDS1, stream>>>(obs, enc_w1, enc_b1, enc_w2, enc_b2,
                                                 out_msgs, ws_msgs);
  k_attn<<<BB, 256, LDS2, stream>>>(wq, bq, wk, bk, wv, bv, wo, bo, ws_msgs, ws_agg);
  k_integrator<<<BB * NN / NN, 256, LDS3, stream>>>(obs, int_w1, int_b1, ln_g, ln_b,
                                                    int_w2, int_b2, ws_agg, out_enriched);
}
